// LSTMfromScratch_77180562309681
// MI455X (gfx1250) — matrix operand order repaired and checked by reading, not compiled, not run on hardware
//
#include <hip/hip_runtime.h>

#define DEVINL __device__ __forceinline__

typedef __attribute__((ext_vector_type(16))) __bf16 v16bf;
typedef __attribute__((ext_vector_type(8)))  float  v8f;

constexpr int B_   = 64;
constexpr int SEQ_ = 512;
constexpr int IN_  = 512;
constexpr int H_   = 1024;
constexpr int G_   = 4 * H_;   // 4096 gate columns
constexpr int OUT_ = 512;
constexpr int COMB_ = IN_ + H_;

union FragU { uint4 u[2]; v16bf v; };

// ---- fragment loaders (ISA 7.12.2 layouts, wave32) -------------------------
// A (16x32 bf16, MxK): lane<16 -> row M=lane, K = [k0..k0+7] then [k0+16..k0+23]
//                      lane>=16 -> row M=lane-16, K = [k0+8..] then [k0+24..]
DEVINL v16bf load_a_frag(const __bf16* base, size_t ld, int row, int k0, int lane) {
  const __bf16* p = base + (size_t)row * ld + k0 + ((lane >> 4) << 3);
  FragU f;
  f.u[0] = *(const uint4*)p;          // 8 bf16: K chunk 0
  f.u[1] = *(const uint4*)(p + 16);   // 8 bf16: K chunk +16
  return f.v;
}
// B (32x16 bf16, KxN) from W^T storage (row = output unit n, contiguous K):
// lane<16 -> col N=lane, K = [k0..k0+7] then [k0+16..k0+23] ; lane>=16 -> col N=lane-16, K = [k0+8..] then [k0+24..]
DEVINL v16bf load_b_frag(const __bf16* base, size_t ld, int nrow, int k0, int lane) {
  const __bf16* p = base + (size_t)nrow * ld + k0 + ((lane >> 4) << 3);
  FragU f;
  f.u[0] = *(const uint4*)p;
  f.u[1] = *(const uint4*)(p + 16);
  return f.v;
}

DEVINL v8f wmma_bf16(v16bf a, v16bf b, v8f c) {
  return __builtin_amdgcn_wmma_f32_16x16x32_bf16(
      /*neg_a=*/false, a, /*neg_b=*/false, b,
      /*c_mod=*/(short)0, c, /*reuse_a=*/false, /*reuse_b=*/false);
}

DEVINL float sigmoidf_(float x) { return 1.0f / (1.0f + __expf(-x)); }

// ---- prep: gate weights f32 [1536,1024] x4 -> WxT bf16 [4096,512], WhT [4096,1024]
__global__ void prep_weights(const float* __restrict__ Wi, const float* __restrict__ Wf,
                             const float* __restrict__ Wc, const float* __restrict__ Wo,
                             __bf16* __restrict__ WxT, __bf16* __restrict__ WhT) {
  size_t total = (size_t)G_ * COMB_;
  for (size_t idx = (size_t)blockIdx.x * blockDim.x + threadIdx.x; idx < total;
       idx += (size_t)gridDim.x * blockDim.x) {
    int n = (int)(idx / COMB_);     // gate-column 0..4095
    int k = (int)(idx % COMB_);     // combined-input row 0..1535
    int g  = n >> 10;
    int n1 = n & (H_ - 1);
    const float* Wg = (g == 0) ? Wi : (g == 1) ? Wf : (g == 2) ? Wc : Wo;
    float v = Wg[(size_t)k * H_ + n1];
    if (k < IN_) WxT[(size_t)n * IN_ + k]        = (__bf16)v;
    else         WhT[(size_t)n * H_ + (k - IN_)] = (__bf16)v;
  }
}

__global__ void f32_to_bf16_k(const float* __restrict__ src, __bf16* __restrict__ dst, size_t n) {
  for (size_t i = (size_t)blockIdx.x * blockDim.x + threadIdx.x; i < n;
       i += (size_t)gridDim.x * blockDim.x)
    dst[i] = (__bf16)src[i];
}

__global__ void init_state(__bf16* __restrict__ h0, __bf16* __restrict__ h1,
                           float* __restrict__ c, float* __restrict__ bias4,
                           const float* __restrict__ bi, const float* __restrict__ bf,
                           const float* __restrict__ bc, const float* __restrict__ bo) {
  int i = blockIdx.x * blockDim.x + threadIdx.x;
  if (i < B_ * H_) {
    h0[i] = (__bf16)0.0f;
    h1[i] = (__bf16)0.0f;
    c[i]  = 0.0f;
  }
  if (i < G_) {
    int g = i >> 10, j = i & (H_ - 1);
    const float* p = (g == 0) ? bi : (g == 1) ? bf : (g == 2) ? bc : bo;
    bias4[i] = p[j];
  }
}

// ---- one LSTM timestep: gates = [x_t, h] @ W + b, cell update ---------------
// grid: 64 WGs (16 hidden units each), 256 threads = 8 waves.
// wave w: gate g=w/2, batch rows (w&1)*32 .. +31 as two 16x16 WMMA tiles.
__global__ __launch_bounds__(256)
void lstm_step(const __bf16* __restrict__ xbf, int t,
               const __bf16* __restrict__ WxT, const __bf16* __restrict__ WhT,
               const float* __restrict__ bias4,
               const __bf16* __restrict__ hprev, __bf16* __restrict__ hnext,
               float* __restrict__ c) {
  __shared__ float lds_g[4][B_][16];

  const int tid   = threadIdx.x;
  const int w     = tid >> 5;
  const int lane  = tid & 31;
  const int g     = w >> 1;
  const int mbase = (w & 1) * 32;
  const int j0    = blockIdx.x * 16;
  const int n     = lane & 15;
  const int m     = lane & 15;
  const int nrow  = g * H_ + j0 + n;

  v8f acc0 = {}, acc1 = {};

  // x_t part: A rows are batch entries of x[b][t][:], row stride = SEQ*IN
  const __bf16* xrow = xbf + (size_t)t * IN_;
  const size_t ldx = (size_t)SEQ_ * IN_;
  for (int k0 = 0; k0 < IN_; k0 += 32) {
    v16bf bfr = load_b_frag(WxT, IN_, nrow, k0, lane);
    v16bf a0  = load_a_frag(xrow, ldx, mbase + m,      k0, lane);
    v16bf a1  = load_a_frag(xrow, ldx, mbase + 16 + m, k0, lane);
    acc0 = wmma_bf16(a0, bfr, acc0);
    acc1 = wmma_bf16(a1, bfr, acc1);
  }

  // h part: A rows are hprev[b][:], row stride = H
  for (int k0 = 0; k0 < H_; k0 += 32) {
    if (k0 + 32 < H_)
      __builtin_prefetch(WhT + (size_t)nrow * H_ + k0 + 32, 0, 1);
    v16bf bfr = load_b_frag(WhT, H_, nrow, k0, lane);
    v16bf a0  = load_a_frag(hprev, H_, mbase + m,      k0, lane);
    v16bf a1  = load_a_frag(hprev, H_, mbase + 16 + m, k0, lane);
    acc0 = wmma_bf16(a0, bfr, acc0);
    acc1 = wmma_bf16(a1, bfr, acc1);
  }

  // add bias, spill gate tiles to LDS (C/D layout: M = r + 8*(lane>=16), N = lane&15)
  const float bias  = bias4[g * H_ + j0 + n];
  const int   mhalf = (lane >> 4) * 8;
#pragma unroll
  for (int r = 0; r < 8; ++r) {
    lds_g[g][mbase + mhalf + r][n]      = acc0[r] + bias;
    lds_g[g][mbase + 16 + mhalf + r][n] = acc1[r] + bias;
  }
  __syncthreads();

  // cell update for the 64x16 hidden slice owned by this workgroup
#pragma unroll
  for (int e = 0; e < 4; ++e) {
    int idx = tid + 256 * e;         // 0..1023
    int b   = idx >> 4;
    int j   = idx & 15;
    float gi = sigmoidf_(lds_g[0][b][j]);
    float gf = sigmoidf_(lds_g[1][b][j]);
    float gc = tanhf(lds_g[2][b][j]);
    float go = sigmoidf_(lds_g[3][b][j]);
    size_t off = (size_t)b * H_ + j0 + j;
    float cn = gf * c[off] + gi * gc;
    c[off] = cn;
    hnext[off] = (__bf16)(go * tanhf(cn));
  }
}

// ---- final FC: out = h_T @ fc_w^T + fc_b  (M=64, N=512, K=1024) -------------
__global__ __launch_bounds__(256)
void fc_out(const __bf16* __restrict__ h, const __bf16* __restrict__ fcT,
            const float* __restrict__ fcb, float* __restrict__ out) {
  const int tid  = threadIdx.x;
  const int w    = tid >> 5;
  const int lane = tid & 31;
  const int tile = blockIdx.x * 8 + w;   // 0..127
  const int mt   = (tile & 3) * 16;
  const int nt   = (tile >> 2) * 16;
  const int m    = lane & 15;
  const int n    = lane & 15;

  v8f acc = {};
  for (int k0 = 0; k0 < H_; k0 += 32) {
    v16bf bfr = load_b_frag(fcT, H_, nt + n, k0, lane);
    v16bf a   = load_a_frag(h,   H_, mt + m, k0, lane);
    acc = wmma_bf16(a, bfr, acc);
  }
  const float bias  = fcb[nt + n];
  const int   mhalf = (lane >> 4) * 8;
#pragma unroll
  for (int r = 0; r < 8; ++r)
    out[(size_t)(mt + mhalf + r) * OUT_ + nt + n] = acc[r] + bias;
}

extern "C" void kernel_launch(void* const* d_in, const int* in_sizes, int n_in,
                              void* d_out, int out_size, void* d_ws, size_t ws_size,
                              hipStream_t stream) {
  (void)in_sizes; (void)n_in; (void)out_size; (void)ws_size;

  const float* x   = (const float*)d_in[0];
  const float* Wi  = (const float*)d_in[1];
  const float* Wf  = (const float*)d_in[2];
  const float* Wc  = (const float*)d_in[3];
  const float* Wo  = (const float*)d_in[4];
  const float* bi  = (const float*)d_in[5];
  const float* bf  = (const float*)d_in[6];
  const float* bc  = (const float*)d_in[7];
  const float* bo  = (const float*)d_in[8];
  const float* fcw = (const float*)d_in[9];
  const float* fcb = (const float*)d_in[10];

  char* ws = (char*)d_ws;
  __bf16* WhT   = (__bf16*)(ws);                    //  8 MB  [4096 x 1024]
  __bf16* WxT   = (__bf16*)(ws + 8388608);          //  4 MB  [4096 x 512]
  __bf16* fcT   = (__bf16*)(ws + 12582912);         //  1 MB  [512 x 1024]
  __bf16* xbf   = (__bf16*)(ws + 13631488);         // 32 MB  [64 x 512 x 512]
  __bf16* h0    = (__bf16*)(ws + 47185920);         // 128 KB
  __bf16* h1    = (__bf16*)(ws + 47316992);         // 128 KB
  float*  c     = (float*)(ws + 47448064);          // 256 KB
  float*  bias4 = (float*)(ws + 47710208);          //  16 KB [4096]

  prep_weights<<<1024, 256, 0, stream>>>(Wi, Wf, Wc, Wo, WxT, WhT);
  f32_to_bf16_k<<<512, 256, 0, stream>>>(fcw, fcT, (size_t)OUT_ * H_);
  f32_to_bf16_k<<<4096, 256, 0, stream>>>(x, xbf, (size_t)B_ * SEQ_ * IN_);
  init_state<<<(B_ * H_ + 255) / 256, 256, 0, stream>>>(h0, h1, c, bias4, bi, bf, bc, bo);

  for (int t = 0; t < SEQ_; ++t) {
    const __bf16* hp = (t & 1) ? h1 : h0;
    __bf16*       hn = (t & 1) ? h0 : h1;
    lstm_step<<<H_ / 16, 256, 0, stream>>>(xbf, t, WxT, WhT, bias4, hp, hn, c);
  }
  // SEQ is even: step t=511 wrote into h0
  fc_out<<<16, 256, 0, stream>>>(h0, fcT, fcb, (float*)d_out);
}
